// GlobalCovPooling_91319594647822
// MI455X (gfx1250) — matrix operand order repaired and checked by reading, not compiled, not run on hardware
//
#include <hip/hip_runtime.h>
#include <hip/hip_bf16.h>
#include <math.h>

// ---------------- problem constants ----------------
#define BSZ   16
#define CDIM  128
#define NPIX  16384           // 128*128
#define SCH   32              // K-split chunks for covariance
#define KSPAN (NPIX / SCH)    // 512
#define KB    64              // k-tile staged in LDS per step
#define XS_STRIDE 72          // halfs per LDS row (16B-aligned rows)

typedef __attribute__((ext_vector_type(16))) _Float16 v16h;
typedef __attribute__((ext_vector_type(8)))  _Float16 v8h;
typedef __attribute__((ext_vector_type(8)))  float    v8f;
typedef __attribute__((ext_vector_type(4)))  float    v4f;
typedef __attribute__((ext_vector_type(2)))  float    v2f;

union frag16 { v16h v; v8h h[2]; };

// =====================================================================
// Kernel 1: partial Gram (f16 WMMA, f32 accum) + partial channel sums.
// One block = (batch b, k-chunk s). 256 threads = 8 waves.
// Wave w computes the 16x128 row-band of the 128x128 Gram partial.
// =====================================================================
__global__ __launch_bounds__(256)
void cov_partial_kernel(const float* __restrict__ x,
                        float* __restrict__ P,      // [B*SCH][128][128]
                        float* __restrict__ mp) {   // [B*SCH][128]
    __shared__ _Float16 Xs[CDIM * XS_STRIDE];       // 18 KB
    __shared__ float meanScratch[CDIM * 16];        // 8 KB

    const int blk  = blockIdx.x;
    const int b    = blk / SCH;
    const int s    = blk % SCH;
    const int t    = threadIdx.x;
    const int wave = t >> 5;
    const int lane = t & 31;
    const int lr   = lane & 15;     // 0..15
    const int lh   = lane >> 4;     // 0..1

    const float* xb = x + (size_t)b * CDIM * NPIX;
    const int kbase0 = s * KSPAN;

    // loader mapping: row r0 = t/16, 4 cols at cq = (t%16)*4 (coalesced float4)
    const int r0 = t >> 4;
    const int cq = (t & 15) * 4;

    float msum[8];
#pragma unroll
    for (int i = 0; i < 8; ++i) msum[i] = 0.f;

    v8f acc[8];
#pragma unroll
    for (int j = 0; j < 8; ++j) acc[j] = (v8f)0.f;

    for (int ch = 0; ch < KSPAN / KB; ++ch) {
        const int kb = kbase0 + ch * KB;
        __syncthreads();   // previous WMMA readers done with LDS
#pragma unroll
        for (int i = 0; i < 8; ++i) {
            const int r = r0 + 16 * i;
            v4f v = *(const v4f*)(xb + (size_t)r * NPIX + kb + cq);
            msum[i] += v.x + v.y + v.z + v.w;
            _Float16* dst = &Xs[r * XS_STRIDE + cq];
            dst[0] = (_Float16)v.x; dst[1] = (_Float16)v.y;
            dst[2] = (_Float16)v.z; dst[3] = (_Float16)v.w;
        }
        __syncthreads();

        // two WMMA k-steps of 32 over the staged 64 columns
#pragma unroll
        for (int ks = 0; ks < 2; ++ks) {
            const int kk = ks * 32;
            // A fragment: rows 16*wave.., per-lane K set {base0..+7, base0+16..+23}
            frag16 af;
            {
                const int ra = wave * 16 + lr;
                const int base0 = kk + lh * 8;
                af.h[0] = *(const v8h*)&Xs[ra * XS_STRIDE + base0];
                af.h[1] = *(const v8h*)&Xs[ra * XS_STRIDE + base0 + 16];
            }
#pragma unroll
            for (int j = 0; j < 8; ++j) {
                // B fragment for sigma = X X^T: B[k, n] = X[n, k]
                frag16 bf;
                const int rb = j * 16 + lr;
                const int bb = kk + lh * 8;
                bf.h[0] = *(const v8h*)&Xs[rb * XS_STRIDE + bb];
                bf.h[1] = *(const v8h*)&Xs[rb * XS_STRIDE + bb + 16];
                acc[j] = __builtin_amdgcn_wmma_f32_16x16x32_f16(
                    false, af.v, false, bf.v, (short)0, acc[j], false, false);
            }
        }
    }

    // write partial Gram (C/D layout: VGPR v -> row v + 8*lh, col lr)
    float* Pout = P + (size_t)blk * (CDIM * CDIM);
#pragma unroll
    for (int j = 0; j < 8; ++j) {
#pragma unroll
        for (int v = 0; v < 8; ++v) {
            const int m = wave * 16 + v + 8 * lh;
            const int n = j * 16 + lr;
            Pout[m * CDIM + n] = acc[j][v];
        }
    }

    // deterministic per-channel partial sums
#pragma unroll
    for (int i = 0; i < 8; ++i)
        meanScratch[(r0 + 16 * i) * 16 + (t & 15)] = msum[i];
    __syncthreads();
    if (t < CDIM) {
        float ssum = 0.f;
#pragma unroll
        for (int q = 0; q < 16; ++q) ssum += meanScratch[t * 16 + q];
        mp[(size_t)blk * CDIM + t] = ssum;
    }
}

// =====================================================================
// Kernel 2: reduce partials -> sigma = G/N - m m^T, trace-normalize,
// write Y0. One block per batch; fixed-order sums (deterministic).
// =====================================================================
__global__ __launch_bounds__(256)
void reduce_sigma_kernel(const float* __restrict__ P,
                         const float* __restrict__ mp,
                         float* __restrict__ Yg) {
    __shared__ float mvec[CDIM];
    __shared__ float dsum[256];

    const int b = blockIdx.x, t = threadIdx.x;

    if (t < CDIM) {
        float s = 0.f;
        for (int ss = 0; ss < SCH; ++ss)
            s += mp[(size_t)(b * SCH + ss) * CDIM + t];
        mvec[t] = s / (float)NPIX;
    }
    __syncthreads();

    float val[64];
    float dlocal = 0.f;
    const float* Pb = P + (size_t)b * SCH * CDIM * CDIM;
#pragma unroll
    for (int i = 0; i < 64; ++i) {
        const int e = t + 256 * i;
        float g = 0.f;
        for (int ss = 0; ss < SCH; ++ss)
            g += Pb[(size_t)ss * CDIM * CDIM + e];
        const int r = e >> 7, c = e & 127;
        const float sv = g / (float)NPIX - mvec[r] * mvec[c];
        val[i] = sv;
        if (r == c) dlocal += sv;
    }
    dsum[t] = dlocal;
    __syncthreads();
    for (int off = 128; off > 0; off >>= 1) {
        if (t < off) dsum[t] += dsum[t + off];
        __syncthreads();
    }
    const float inv_tr = 1.f / dsum[0];
#pragma unroll
    for (int i = 0; i < 64; ++i) {
        const int e = t + 256 * i;
        Yg[(size_t)b * CDIM * CDIM + e] = val[i] * inv_tr;
    }
}

// =====================================================================
// Kernel 3: persistent Newton-Schulz, one block per batch, f32 WMMA
// (v_wmma_f32_16x16x4_f32). Y/Z/T live in L2-resident workspace.
// Tail: trace-compensated column means -> out[b][c].
// =====================================================================
__global__ __launch_bounds__(256)
void ns_kernel(float* __restrict__ Yg, float* __restrict__ Zg,
               float* __restrict__ Tg, float* __restrict__ out) {
    __shared__ float sred[CDIM + 1];
    __shared__ float scol[CDIM];

    const int b = blockIdx.x, t = threadIdx.x;
    const int wave = t >> 5, lane = t & 31;
    const int lr = lane & 15, lh = lane >> 4;

    float* Y = Yg + (size_t)b * CDIM * CDIM;
    float* Z = Zg + (size_t)b * CDIM * CDIM;
    float* T = Tg + (size_t)b * CDIM * CDIM;

    // Z = I
    for (int e = t; e < CDIM * CDIM; e += 256) {
        const int r = e >> 7, c = e & 127;
        Z[e] = (r == c) ? 1.f : 0.f;
    }
    __syncthreads();

    for (int it = 0; it < 6; ++it) {
        // ---- phase A: acc = Z @ Y  (wave w -> row band [16w,16w+16)) ----
        v8f acc[8];
#pragma unroll
        for (int j = 0; j < 8; ++j) acc[j] = (v8f)0.f;
        for (int k0 = 0; k0 < CDIM; k0 += 4) {
            v2f a;
            const int ra = wave * 16 + lr;
            const int ka = k0 + 2 * lh;            // f32 A layout: lanes<16 K=0,1; >=16 K=2,3
            a.x = Z[ra * CDIM + ka];
            a.y = Z[ra * CDIM + ka + 1];
#pragma unroll
            for (int j = 0; j < 8; ++j) {
                v2f bb;
                const int n = j * 16 + lr;
                bb.x = Y[ka * CDIM + n];
                bb.y = Y[(ka + 1) * CDIM + n];
                acc[j] = __builtin_amdgcn_wmma_f32_16x16x4_f32(
                    false, a, false, bb, (short)0, acc[j], false, false);
            }
        }
        __syncthreads();   // all Y/Z reads complete

        // ---- phase B: T = 3I - ZY ; Y = 0.5 * Y .* T (elementwise) ----
#pragma unroll
        for (int j = 0; j < 8; ++j) {
#pragma unroll
            for (int v = 0; v < 8; ++v) {
                const int m = wave * 16 + v + 8 * lh;
                const int n = j * 16 + lr;
                const float tv = ((m == n) ? 3.f : 0.f) - acc[j][v];
                T[m * CDIM + n] = tv;
                Y[m * CDIM + n] = Y[m * CDIM + n] * 0.5f * tv;
            }
        }
        __syncthreads();   // T visible, Y updated

        // ---- phase C: acc = T @ Z (scaled 0.5 at store) ----
#pragma unroll
        for (int j = 0; j < 8; ++j) acc[j] = (v8f)0.f;
        for (int k0 = 0; k0 < CDIM; k0 += 4) {
            v2f a;
            const int ra = wave * 16 + lr;
            const int ka = k0 + 2 * lh;
            a.x = T[ra * CDIM + ka];
            a.y = T[ra * CDIM + ka + 1];
#pragma unroll
            for (int j = 0; j < 8; ++j) {
                v2f bb;
                const int n = j * 16 + lr;
                bb.x = Z[ka * CDIM + n];
                bb.y = Z[(ka + 1) * CDIM + n];
                acc[j] = __builtin_amdgcn_wmma_f32_16x16x4_f32(
                    false, a, false, bb, (short)0, acc[j], false, false);
            }
        }
        __syncthreads();   // all Z reads complete before overwrite

        // ---- phase D: Z = 0.5 * acc ----
#pragma unroll
        for (int j = 0; j < 8; ++j) {
#pragma unroll
            for (int v = 0; v < 8; ++v) {
                const int m = wave * 16 + v + 8 * lh;
                const int n = j * 16 + lr;
                Z[m * CDIM + n] = 0.5f * acc[j][v];
            }
        }
        __syncthreads();
    }

    // ---- tail: out[b][c] = sqrt(tr(Y)) * mean_rows(Y)[c] ----
    if (t < CDIM) {
        float cs = 0.f;
        for (int r = 0; r < CDIM; ++r) cs += Y[r * CDIM + t];
        scol[t] = cs;
        sred[t] = Y[t * CDIM + t];
    }
    __syncthreads();
    if (t == 0) {
        float tr = 0.f;
        for (int i = 0; i < CDIM; ++i) tr += sred[i];
        sred[CDIM] = sqrtf(tr);
    }
    __syncthreads();
    if (t < CDIM)
        out[b * CDIM + t] = sred[CDIM] * scol[t] * (1.f / (float)CDIM);
}

// =====================================================================
extern "C" void kernel_launch(void* const* d_in, const int* in_sizes, int n_in,
                              void* d_out, int out_size, void* d_ws, size_t ws_size,
                              hipStream_t stream) {
    const float* x = (const float*)d_in[0];
    float* out = (float*)d_out;

    char* ws = (char*)d_ws;
    size_t off = 0;
    float* P  = (float*)(ws + off); off += (size_t)BSZ * SCH * CDIM * CDIM * sizeof(float); // 32 MB
    float* mp = (float*)(ws + off); off += (size_t)BSZ * SCH * CDIM * sizeof(float);        // 256 KB
    float* Yg = (float*)(ws + off); off += (size_t)BSZ * CDIM * CDIM * sizeof(float);       // 1 MB
    float* Zg = (float*)(ws + off); off += (size_t)BSZ * CDIM * CDIM * sizeof(float);       // 1 MB
    float* Tg = (float*)(ws + off); off += (size_t)BSZ * CDIM * CDIM * sizeof(float);       // 1 MB
    (void)ws_size; (void)in_sizes; (void)n_in; (void)out_size;

    cov_partial_kernel<<<BSZ * SCH, 256, 0, stream>>>(x, P, mp);
    reduce_sigma_kernel<<<BSZ, 256, 0, stream>>>(P, mp, Yg);
    ns_kernel<<<BSZ, 256, 0, stream>>>(Yg, Zg, Tg, out);
}
